// RRNNModel_17772574671609
// MI455X (gfx1250) — matrix operand order repaired and checked by reading, not compiled, not run on hardware
//
#include <hip/hip_runtime.h>
#include <hip/hip_bf16.h>

typedef __attribute__((ext_vector_type(16))) __bf16 v16bf;
typedef __attribute__((ext_vector_type(8)))  __bf16 v8bf;
typedef __attribute__((ext_vector_type(8)))  float  v8f;

#define BS    32
#define SEQ   512
#define VOCAB 5000
#define EMS   256
#define NHID  512
#define G4    (4*NHID)            // 2048 gate width
#define DECNT ((VOCAB + 15)/16)   // 313 N-tiles for decode
#define DECNG ((DECNT + 3)/4)     // 79 groups of 4 N-tiles

// ---------------------------------------------------------------------------
// WMMA fragment loaders (bf16, 16x16x32) following CDNA5 ISA §7.12.2 layouts.
// A: lane L holds row M=L%16; kb=L>>4 selects K chunks {kb*8.., 16+kb*8..}
// B: lane L holds col N=L%16; same K chunks as A {kb*8.., 16+kb*8..}.
// ---------------------------------------------------------------------------
__device__ __forceinline__ v16bf load_frag_a(const __bf16* rowptr, int k0, int kb) {
    v8bf lo = *(const v8bf*)(rowptr + k0 + kb * 8);
    v8bf hi = *(const v8bf*)(rowptr + k0 + 16 + kb * 8);
    return __builtin_shufflevector(lo, hi, 0,1,2,3,4,5,6,7,8,9,10,11,12,13,14,15);
}
__device__ __forceinline__ v16bf load_frag_b(const __bf16* rowptr, int k0, int kb) {
    v8bf lo = *(const v8bf*)(rowptr + k0 + kb * 8);
    v8bf hi = *(const v8bf*)(rowptr + k0 + 16 + kb * 8);
    return __builtin_shufflevector(lo, hi, 0,1,2,3,4,5,6,7,8,9,10,11,12,13,14,15);
}
__device__ __forceinline__ float sigf(float x) { return 1.0f / (1.0f + __expf(-x)); }

// ---------------------------------------------------------------------------
// Kernel 0: fp32 -> bf16 weight/embedding conversion (grid-stride).
// ---------------------------------------------------------------------------
__global__ void cvt_bf16(const float* __restrict__ src, __bf16* __restrict__ dst, int n) {
    int i = blockIdx.x * blockDim.x + threadIdx.x;
    int stride = gridDim.x * blockDim.x;
    for (; i < n; i += stride) dst[i] = (__bf16)src[i];
}

// ---------------------------------------------------------------------------
// Kernel 1: Xg[t*32+b][n] = emb[inputs[b,t]] @ W_ih^T + (b_ih + b_hh)
// M=16384 (r = t*32+b), N=2048, K=256. 1x4 N-blocked tiles per wave:
// one A fragment + 4 pre-staged B fragments per K-step.
// ---------------------------------------------------------------------------
__global__ void xgate_gemm(const __bf16* __restrict__ embB,
                           const __bf16* __restrict__ W_ihB,
                           const int*    __restrict__ inputs,
                           const float*  __restrict__ b_ih,
                           const float*  __restrict__ b_hh,
                           float* __restrict__ Xg) {
    int wid  = blockIdx.x * (blockDim.x >> 5) + (threadIdx.x >> 5);
    int lane = threadIdx.x & 31;
    int lcol = lane & 15;
    int kb   = lane >> 4;
    const int NG = G4 / 64;     // 32 groups of 4 N-tiles
    int mt = wid / NG;          // 0..1023
    int ng = wid % NG;          // 0..31

    int r   = mt * 16 + lcol;   // r = t*32 + b
    int t   = r >> 5, b = r & 31;
    int tok = inputs[b * SEQ + t];
    const __bf16* arow = embB + (size_t)tok * EMS;

    int n[4];
    const __bf16* brow[4];
    #pragma unroll
    for (int j = 0; j < 4; ++j) {
        n[j] = (ng * 4 + j) * 16 + lcol;
        brow[j] = W_ihB + (size_t)n[j] * EMS;
    }

    v8f acc[4] = {};
    #pragma unroll
    for (int k0 = 0; k0 < EMS; k0 += 32) {
        v16bf af = load_frag_a(arow, k0, kb);
        v16bf bf[4];
        #pragma unroll
        for (int j = 0; j < 4; ++j) bf[j] = load_frag_b(brow[j], k0, kb);
        #pragma unroll
        for (int j = 0; j < 4; ++j)
            acc[j] = __builtin_amdgcn_wmma_f32_16x16x32_bf16(false, af, false, bf[j],
                                                             (short)0, acc[j], false, false);
    }
    int hi8 = kb * 8;
    #pragma unroll
    for (int j = 0; j < 4; ++j) {
        float bias = b_ih[n[j]] + b_hh[n[j]];
        #pragma unroll
        for (int e = 0; e < 8; ++e) {
            int row = mt * 16 + e + hi8;
            Xg[(size_t)row * G4 + n[j]] = acc[j][e] + bias;
        }
    }
}

// ---------------------------------------------------------------------------
// Kernel 2: sequential LSTM scan. One persistent 1024-thread workgroup;
// h in LDS (bf16, padded rows), c in accumulator registers. A fragment (h)
// loaded once per K-step; 4 gate B fragments pre-staged so WMMAs overlap
// the L2 loads on the serial critical path.
// ---------------------------------------------------------------------------
__global__ void __launch_bounds__(1024, 1)
lstm_scan(const __bf16* __restrict__ W_hhB,
          const float*  __restrict__ Xg,
          const float*  __restrict__ h0,
          const float*  __restrict__ c0,
          __bf16* __restrict__ hsB,     // [b*SEQ + t][NHID] bf16 hidden states
          float* __restrict__ hT, float* __restrict__ cT) {
    __shared__ __bf16 hlds[BS][NHID + 8];   // +8 pad: avoid bank conflicts

    int tid  = threadIdx.x;
    int wid  = tid >> 5;          // wave = n'-tile, 0..31
    int lane = tid & 31;
    int lcol = lane & 15;
    int kb   = lane >> 4;
    int hi8  = kb * 8;
    int nb   = wid * 16;

    for (int i = tid; i < BS * NHID; i += 1024)
        hlds[i >> 9][i & 511] = (__bf16)h0[i];

    v8f cst[2];
    #pragma unroll
    for (int bt = 0; bt < 2; ++bt)
        #pragma unroll
        for (int e = 0; e < 8; ++e) {
            int b = bt * 16 + e + hi8;
            cst[bt][e] = c0[b * NHID + nb + lcol];
        }

    const __bf16* brow[4];
    #pragma unroll
    for (int g = 0; g < 4; ++g)
        brow[g] = W_hhB + (size_t)(g * NHID + nb + lcol) * NHID;

    __syncthreads();

    for (int t = 0; t < SEQ; ++t) {
        v8f hn[2];
        #pragma unroll
        for (int bt = 0; bt < 2; ++bt) {
            int mb = bt * 16;
            const __bf16* arow = &hlds[mb + lcol][0];
            v8f acc[4] = {};
            #pragma unroll
            for (int k0 = 0; k0 < NHID; k0 += 32) {
                v16bf af = load_frag_a(arow, k0, kb);   // shared across 4 gates
                v16bf bf[4];
                #pragma unroll
                for (int g = 0; g < 4; ++g) bf[g] = load_frag_b(brow[g], k0, kb);
                #pragma unroll
                for (int g = 0; g < 4; ++g)
                    acc[g] = __builtin_amdgcn_wmma_f32_16x16x32_bf16(false, af, false, bf[g],
                                                                     (short)0, acc[g], false, false);
            }
            #pragma unroll
            for (int g = 0; g < 4; ++g)
                #pragma unroll
                for (int e = 0; e < 8; ++e) {
                    int row = t * 32 + mb + e + hi8;
                    acc[g][e] += Xg[(size_t)row * G4 + (g * NHID + nb + lcol)];
                }
            #pragma unroll
            for (int e = 0; e < 8; ++e) {
                float iv = sigf(acc[0][e]);
                float fv = sigf(acc[1][e]);
                float gv = tanhf(acc[2][e]);
                float ov = sigf(acc[3][e]);
                float cn = fv * cst[bt][e] + iv * gv;
                cst[bt][e] = cn;
                hn[bt][e]  = ov * tanhf(cn);
            }
        }
        __syncthreads();   // all reads of hlds done for this step
        #pragma unroll
        for (int bt = 0; bt < 2; ++bt)
            #pragma unroll
            for (int e = 0; e < 8; ++e) {
                int b = bt * 16 + e + hi8;
                int n = nb + lcol;
                __bf16 hb = (__bf16)hn[bt][e];
                hlds[b][n] = hb;
                hsB[((size_t)b * SEQ + t) * NHID + n] = hb;
            }
        __syncthreads();   // hlds ready for next step
        if (t == SEQ - 1) {
            #pragma unroll
            for (int bt = 0; bt < 2; ++bt)
                #pragma unroll
                for (int e = 0; e < 8; ++e) {
                    int b = bt * 16 + e + hi8;
                    int n = nb + lcol;
                    hT[b * NHID + n] = hn[bt][e];
                    cT[b * NHID + n] = cst[bt][e];
                }
        }
    }
}

// ---------------------------------------------------------------------------
// Kernel 3: decode GEMM  dec[r][n] = hs[r] . W_dec[n] + b_dec[n]
// M=16384 (r=b*512+t), N=5000, K=512. 1x4 N-blocked tiles per wave; one A
// fragment + 4 pre-staged B fragments per K-step. Edge N-tiles: clamped
// loads, masked stores (EXEC divergence only outside the WMMA loop).
// ---------------------------------------------------------------------------
__global__ void decode_gemm(const __bf16* __restrict__ hsB,
                            const __bf16* __restrict__ W_decB,
                            const float*  __restrict__ b_dec,
                            float* __restrict__ dec) {
    int wid = blockIdx.x * (blockDim.x >> 5) + (threadIdx.x >> 5);
    const int total = (BS * SEQ / 16) * DECNG;      // 1024 * 79
    if (wid >= total) return;          // wave-uniform: EXEC full for WMMA
    int lane = threadIdx.x & 31;
    int lcol = lane & 15;
    int kb   = lane >> 4;
    int hi8  = kb * 8;
    int mt = wid / DECNG, ng = wid % DECNG;

    int r = mt * 16 + lcol;
    const __bf16* arow = hsB + (size_t)r * NHID;

    int n[4], nc[4];
    const __bf16* brow[4];
    #pragma unroll
    for (int j = 0; j < 4; ++j) {
        n[j]  = (ng * 4 + j) * 16 + lcol;
        nc[j] = n[j] < VOCAB ? n[j] : VOCAB - 1;    // clamp loads on edge tiles
        brow[j] = W_decB + (size_t)nc[j] * NHID;
    }

    v8f acc[4] = {};
    #pragma unroll
    for (int k0 = 0; k0 < NHID; k0 += 32) {
        v16bf af = load_frag_a(arow, k0, kb);       // shared across 4 N-tiles
        v16bf bf[4];
        #pragma unroll
        for (int j = 0; j < 4; ++j) bf[j] = load_frag_b(brow[j], k0, kb);
        #pragma unroll
        for (int j = 0; j < 4; ++j)
            acc[j] = __builtin_amdgcn_wmma_f32_16x16x32_bf16(false, af, false, bf[j],
                                                             (short)0, acc[j], false, false);
    }
    #pragma unroll
    for (int j = 0; j < 4; ++j) {
        float bias = b_dec[nc[j]];
        if (n[j] < VOCAB) {
            #pragma unroll
            for (int e = 0; e < 8; ++e) {
                int row = mt * 16 + e + hi8;
                dec[(size_t)row * VOCAB + n[j]] = acc[j][e] + bias;
            }
        }
    }
}

// ---------------------------------------------------------------------------
extern "C" void kernel_launch(void* const* d_in, const int* in_sizes, int n_in,
                              void* d_out, int out_size, void* d_ws, size_t ws_size,
                              hipStream_t stream) {
    (void)in_sizes; (void)n_in; (void)out_size; (void)ws_size;
    const int*   inputs = (const int*)  d_in[0];
    const float* h0     = (const float*)d_in[1];
    const float* c0     = (const float*)d_in[2];
    // d_in[3] measure_weights, d_in[4] measure_steps: dead code (GAMMA == 0)
    const float* emb    = (const float*)d_in[5];
    const float* W_ih   = (const float*)d_in[6];
    const float* W_hh   = (const float*)d_in[7];
    const float* b_ih   = (const float*)d_in[8];
    const float* b_hh   = (const float*)d_in[9];
    const float* W_dec  = (const float*)d_in[10];
    const float* b_dec  = (const float*)d_in[11];

    float* out = (float*)d_out;
    float* dec = out;                                  // [32][512][5000]
    float* hT  = out + (size_t)BS * SEQ * VOCAB;       // [32][512]
    float* cT  = hT + (size_t)BS * NHID;               // [32][512]

    char* ws = (char*)d_ws;
    size_t o = 0;
    auto carve = [&](size_t bytes) -> char* {
        char* p = ws + o;
        o += (bytes + 255) & ~(size_t)255;
        return p;
    };
    __bf16* embB   = (__bf16*)carve((size_t)VOCAB * EMS  * 2);
    __bf16* W_ihB  = (__bf16*)carve((size_t)G4    * EMS  * 2);
    __bf16* W_hhB  = (__bf16*)carve((size_t)G4    * NHID * 2);
    __bf16* W_decB = (__bf16*)carve((size_t)VOCAB * NHID * 2);
    float*  Xg     = (float*) carve((size_t)SEQ * BS * G4 * 4);  // 134 MB
    __bf16* hsB    = (__bf16*)carve((size_t)BS * SEQ * NHID * 2);

    cvt_bf16<<<512, 256, 0, stream>>>(emb,   embB,   VOCAB * EMS);
    cvt_bf16<<<512, 256, 0, stream>>>(W_ih,  W_ihB,  G4 * EMS);
    cvt_bf16<<<512, 256, 0, stream>>>(W_hh,  W_hhB,  G4 * NHID);
    cvt_bf16<<<512, 256, 0, stream>>>(W_dec, W_decB, VOCAB * NHID);

    // 1024 M-tiles x 32 N-groups = 32768 waves, 8 waves/block
    xgate_gemm<<<4096, 256, 0, stream>>>(embB, W_ihB, inputs, b_ih, b_hh, Xg);

    // serial scan: single WGP-resident workgroup
    lstm_scan<<<1, 1024, 0, stream>>>(W_hhB, Xg, h0, c0, hsB, hT, cT);

    // 1024 M-tiles x 79 N-groups = 80896 waves, 8 waves/block
    decode_gemm<<<(80896 + 7) / 8, 256, 0, stream>>>(hsB, W_decB, b_dec, dec);
}
